// MSFusionV1_74517682585708
// MI455X (gfx1250) — matrix operand order repaired and checked by reading, not compiled, not run on hardware
//
#include <hip/hip_runtime.h>
#include <hip/hip_bf16.h>

typedef __attribute__((ext_vector_type(16))) __bf16 v16bf;
typedef __attribute__((ext_vector_type(8)))  float  v8f;

#define C1 64
#define HH 256
#define WW 256

static __device__ __forceinline__ v8f wmma_bf16(v16bf a, v16bf b, v8f c) {
    return __builtin_amdgcn_wmma_f32_16x16x32_bf16(false, a, false, b,
                                                   (short)0, c, false, false);
}

// A-matrix (16x32 bf16) element map: lane -> row m=lane&15, half=lane>>4;
// element e -> K = half*16+e (this code's contiguous order; B uses the same).
static __device__ __forceinline__ int a_k_of(int e, int half) {
    return half * 16 + e;
}

// ---------------- Kernel 0: repack weights into WMMA A-fragment order -------
// Layouts (bf16): gwF/lwF: [ob(4)][kk(2)][lane(32)][e(16)]
//                 cwF    : [ob(4)][kk(18)][lane(32)][e(16)], K = tap*64 + c
__global__ void repack_kernel(const float* __restrict__ gw,
                              const float* __restrict__ lw,
                              const float* __restrict__ cw,
                              __bf16* __restrict__ gwF,
                              __bf16* __restrict__ lwF,
                              __bf16* __restrict__ cwF) {
    for (int i = threadIdx.x; i < 4 * 2 * 32 * 16; i += 256) {
        int e = i & 15, lane = (i >> 4) & 31, kk = (i >> 9) & 1, ob = i >> 10;
        int m = lane & 15, half = lane >> 4;
        int o = ob * 16 + m, c = kk * 32 + a_k_of(e, half);
        gwF[i] = (__bf16)gw[o * C1 + c];
        lwF[i] = (__bf16)lw[o * C1 + c];
    }
    for (int i = threadIdx.x; i < 4 * 18 * 32 * 16; i += 256) {
        int e = i & 15, lane = (i >> 4) & 31;
        int kk = (i >> 9) % 18, ob = (i >> 9) / 18;
        int m = lane & 15, half = lane >> 4;
        int K2 = kk * 32 + a_k_of(e, half);
        int tap = K2 >> 6, c = K2 & 63, o = ob * 16 + m;
        cwF[i] = (__bf16)cw[(o * C1 + c) * 9 + tap];
    }
}

// ---------------- Kernel 1a: pooled = mean over H,W of g --------------------
__global__ void pooled_kernel(const float* __restrict__ x, float* __restrict__ pooled) {
    const int c = blockIdx.x, b = blockIdx.y;
    const float* p = x + ((size_t)(b * 5 * C1 + c) << 16);
    float s = 0.f;
    for (int i = threadIdx.x; i < HH * WW; i += 256) s += p[i];
    __shared__ float red[256];
    red[threadIdx.x] = s;
    __syncthreads();
    for (int off = 128; off > 0; off >>= 1) {
        if ((int)threadIdx.x < off) red[threadIdx.x] += red[threadIdx.x + off];
        __syncthreads();
    }
    if (threadIdx.x == 0) pooled[b * C1 + c] = red[0] * (1.f / (HH * WW));
}

// ---------------- Kernel 1b: x_mean = mean over channels of g ---------------
__global__ void xmean_kernel(const float* __restrict__ x, float* __restrict__ xmean) {
    const int gid = blockIdx.x * 256 + threadIdx.x;
    const int b = gid >> 16, p = gid & 65535;
    const float* base = x + (((size_t)b * 5 * C1) << 16) + p;
    float s = 0.f;
    #pragma unroll 8
    for (int c = 0; c < C1; ++c) {
        if (c + 8 < C1) __builtin_prefetch(base + ((size_t)(c + 8) << 16), 0, 0);
        s += base[(size_t)c << 16];
    }
    xmean[gid] = s * (1.f / C1);
}

// ---------------- Kernel 2: tiny MLP -> per-sample 3x3 kernel ---------------
__global__ void mlp_kernel(const float* __restrict__ pooled,
                           const float* __restrict__ w1, const float* __restrict__ b1,
                           const float* __restrict__ w2, const float* __restrict__ b2,
                           float* __restrict__ kern) {
    __shared__ float hp[C1];
    __shared__ float pl[C1];
    const int t = threadIdx.x;
    for (int b = 0; b < 4; ++b) {
        pl[t] = pooled[b * C1 + t];
        __syncthreads();
        float acc = b1[t];
        #pragma unroll 8
        for (int c = 0; c < C1; ++c) acc += w1[t * C1 + c] * pl[c];
        hp[t] = fmaxf(acc, 0.f);
        __syncthreads();
        if (t < 9) {
            float k = b2[t];
            #pragma unroll 8
            for (int c = 0; c < C1; ++c) k += w2[t * C1 + c] * hp[c];
            kern[b * 9 + t] = k;
        }
        __syncthreads();
    }
}

// ------- Kernel 3: att = sigmoid(conv3x3(x_mean,k)); proj_global GEMM -------
__global__ void attn_projg_kernel(const float* __restrict__ x,
                                  const float* __restrict__ xmean,
                                  const float* __restrict__ kern,
                                  const __bf16* __restrict__ gwF,
                                  const float* __restrict__ gb,
                                  float* __restrict__ projg) {
    const int b  = blockIdx.y;
    const int p0 = blockIdx.x * 32;
    const int y  = p0 >> 8;
    const int x0 = p0 & 255;
    const int tid = threadIdx.x;

    __shared__ float attS[32];
    // B fragments in-layout: [nb(2)][kk(2)][lane(32)][e(16)]
    __shared__ __attribute__((aligned(32))) __bf16 actF[2][2][32][16];

    if (tid < 32) {
        float k[9];
        #pragma unroll
        for (int i = 0; i < 9; ++i) k[i] = kern[b * 9 + i];
        const float* xm = xmean + ((size_t)b << 16);
        const int xc = x0 + tid;
        float s = 0.f;
        #pragma unroll
        for (int dy = -1; dy <= 1; ++dy)
            #pragma unroll
            for (int dx = -1; dx <= 1; ++dx) {
                int yy = y + dy, xx = xc + dx;
                float v = (yy >= 0 && yy < HH && xx >= 0 && xx < WW)
                              ? xm[yy * WW + xx] : 0.f;
                s += v * k[(dy + 1) * 3 + (dx + 1)];
            }
        attS[tid] = 1.f / (1.f + __expf(-s));
    }
    __syncthreads();
    // activations = g * att, written directly in B-fragment order
    for (int i = tid; i < C1 * 32; i += 256) {
        int c = i >> 5, j = i & 31;
        float v = x[(((size_t)(b * 5 * C1 + c)) << 16) + p0 + j] * attS[j];
        int kk = c >> 5, Kc = c & 31;
        actF[j >> 4][kk][(j & 15) + ((Kc >> 4) << 4)][Kc & 15] = (__bf16)v;
    }
    __syncthreads();

    const int wv = tid >> 5, lane = tid & 31;
    const int ob = wv >> 1, nb = wv & 1;
    const int m = lane & 15, half = lane >> 4;
    const v16bf* aG = reinterpret_cast<const v16bf*>(gwF) + (ob * 2) * 32 + lane;
    v8f acc = {};
    #pragma unroll
    for (int kk = 0; kk < 2; ++kk) {
        v16bf a  = aG[kk * 32];
        v16bf bf = *reinterpret_cast<const v16bf*>(&actF[nb][kk][lane][0]);
        acc = wmma_bf16(a, bf, acc);
    }
    #pragma unroll
    for (int v = 0; v < 8; ++v) {
        int o   = ob * 16 + half * 8 + v;
        int pix = p0 + nb * 16 + m;
        projg[(((size_t)(b * C1 + o)) << 16) + pix] = acc[v] + gb[o];
    }
}

// ---- Kernel 4: proj_local GEMM + bilinear gate + conv3x3 s2 GEMM ----------
// Output tile 4x8 @256 res; fused halo 9x17 @512 res.
__global__ void fuse_conv_kernel(const float* __restrict__ x,
                                 const float* __restrict__ projg,
                                 const __bf16* __restrict__ lwF,
                                 const float* __restrict__ lb,
                                 const float* __restrict__ fwp,
                                 const __bf16* __restrict__ cwF,
                                 const float* __restrict__ bng,
                                 const float* __restrict__ bnb,
                                 const float* __restrict__ bnm,
                                 const float* __restrict__ bnv,
                                 float* __restrict__ out) {
    const int b  = blockIdx.z;
    const int Y0 = blockIdx.y * 4;
    const int X0 = blockIdx.x * 8;
    const int tid = threadIdx.x;

    // local activations in B-fragment order: [jb(10)][kk(2)][lane(32)][e(16)]
    __shared__ __attribute__((aligned(32))) __bf16 locF[10][2][32][16];
    // fused activations transposed (j, c): conv B-fragments are contiguous
    __shared__ __attribute__((aligned(32))) __bf16 fusedT[160][C1];

    for (int i = tid; i < C1 * 160; i += 256) {
        int c = i / 160, j = i % 160;
        float v = 0.f;
        if (j < 153) {
            int fy = 2 * Y0 - 1 + j / 17;
            int fx = 2 * X0 - 1 + j % 17;
            if (fy >= 0 && fx >= 0) {
                int quad = ((fy >> 8) << 1) | (fx >> 8);
                int grp  = C1 * (1 + quad) + c;
                v = x[(((size_t)(b * 5 * C1 + grp)) << 16)
                      + ((fy & 255) << 8) + (fx & 255)];
            }
        }
        int kk = c >> 5, Kc = c & 31;
        locF[j >> 4][kk][(j & 15) + ((Kc >> 4) << 4)][Kc & 15] = (__bf16)v;
    }
    __syncthreads();

    const float alpha = 1.f / (1.f + __expf(-fwp[0]));
    const int wv = tid >> 5, lane = tid & 31;
    const int m = lane & 15, half = lane >> 4;

    // ---- GEMM1: proj_local (K=64), 4 c-blocks x 10 j-blocks ----
    for (int t = wv; t < 40; t += 8) {
        int cb = t / 10, jb = t % 10;
        v8f acc = {};
        #pragma unroll
        for (int kk = 0; kk < 2; ++kk) {
            v16bf a  = *(reinterpret_cast<const v16bf*>(lwF) + (cb * 2 + kk) * 32 + lane);
            v16bf bf = *reinterpret_cast<const v16bf*>(&locF[jb][kk][lane][0]);
            acc = wmma_bf16(a, bf, acc);
        }
        #pragma unroll
        for (int v = 0; v < 8; ++v) {
            int c = cb * 16 + half * 8 + v;
            int j = jb * 16 + m;
            if (j < 153) {
                float f = 0.f;
                int fy = 2 * Y0 - 1 + j / 17;
                int fx = 2 * X0 - 1 + j % 17;
                if (fy >= 0 && fx >= 0) {
                    float sy = fy * 0.5f - 0.25f;
                    float sx = fx * 0.5f - 0.25f;
                    int iy = (int)floorf(sy), ix = (int)floorf(sx);
                    float wy = sy - (float)iy, wx = sx - (float)ix;
                    int y0c = iy < 0 ? 0 : iy;
                    int y1c = iy + 1 > 255 ? 255 : iy + 1;
                    int x0c = ix < 0 ? 0 : ix;
                    int x1c = ix + 1 > 255 ? 255 : ix + 1;
                    const float* pg = projg + (((size_t)(b * C1 + c)) << 16);
                    float v00 = pg[y0c * 256 + x0c], v01 = pg[y0c * 256 + x1c];
                    float v10 = pg[y1c * 256 + x0c], v11 = pg[y1c * 256 + x1c];
                    float up = (1.f - wy) * ((1.f - wx) * v00 + wx * v01)
                             +        wy  * ((1.f - wx) * v10 + wx * v11);
                    f = alpha * up + (1.f - alpha) * (acc[v] + lb[c]);
                }
                fusedT[j][c] = (__bf16)f;
            }
        }
    }
    __syncthreads();

    // ---- GEMM2: conv3x3 s2 as K=576 GEMM (K = tap*64 + c, tap-major) ----
    const int ob = wv >> 1, nb = wv & 1;
    const int t2 = nb * 16 + m;
    const int by = 2 * (t2 >> 3), bx = 2 * (t2 & 7);
    const v16bf* aC = reinterpret_cast<const v16bf*>(cwF) + (ob * 18) * 32 + lane;
    v8f acc2 = {};
    #pragma unroll
    for (int kk = 0; kk < 18; ++kk) {
        v16bf a = aC[kk * 32];
        int K2 = kk * 32 + half * 16;     // fragment run stays inside one tap
        int tap = K2 >> 6, c0 = K2 & 63;
        int dy = tap / 3, dx = tap % 3;
        int j = (by + dy) * 17 + bx + dx;
        v16bf bf = *reinterpret_cast<const v16bf*>(&fusedT[j][c0]);
        acc2 = wmma_bf16(a, bf, acc2);
    }
    #pragma unroll
    for (int v = 0; v < 8; ++v) {
        int o = ob * 16 + half * 8 + v;
        int t = nb * 16 + m;
        int oy = Y0 + (t >> 3), ox = X0 + (t & 7);
        float scale = bng[o] * rsqrtf(bnv[o] + 1e-3f);
        float yv = acc2[v] * scale + (bnb[o] - bnm[o] * scale);
        out[(((size_t)(b * C1 + o)) << 16) + (oy << 8) + ox] =
            yv / (1.f + __expf(-yv));
    }
}

extern "C" void kernel_launch(void* const* d_in, const int* in_sizes, int n_in,
                              void* d_out, int out_size, void* d_ws, size_t ws_size,
                              hipStream_t stream) {
    const float* x       = (const float*)d_in[0];
    const float* dsa_w1  = (const float*)d_in[1];
    const float* dsa_b1  = (const float*)d_in[2];
    const float* dsa_w2  = (const float*)d_in[3];
    const float* dsa_b2  = (const float*)d_in[4];
    const float* gff_gw  = (const float*)d_in[5];
    const float* gff_gb  = (const float*)d_in[6];
    const float* gff_lw  = (const float*)d_in[7];
    const float* gff_lb  = (const float*)d_in[8];
    const float* fweight = (const float*)d_in[9];
    const float* conv_w  = (const float*)d_in[10];
    const float* bn_g    = (const float*)d_in[11];
    const float* bn_b    = (const float*)d_in[12];
    const float* bn_m    = (const float*)d_in[13];
    const float* bn_v    = (const float*)d_in[14];

    float* ws     = (float*)d_ws;
    float* pooled = ws;                    // 256 floats
    float* kern   = ws + 256;              // 36 floats (padded)
    float* xmean  = ws + 512;              // 4*65536 floats
    float* projg  = ws + 512 + 4 * 65536;  // 4*64*65536 floats (~64 MB)
    __bf16* frag  = (__bf16*)(ws + 512 + 4 * 65536 + (size_t)4 * C1 * 65536);
    __bf16* gwF = frag;                    // 4096 bf16
    __bf16* lwF = frag + 4096;             // 4096 bf16
    __bf16* cwF = frag + 8192;             // 36864 bf16

    repack_kernel<<<1, 256, 0, stream>>>(gff_gw, gff_lw, conv_w, gwF, lwF, cwF);
    pooled_kernel<<<dim3(C1, 4), 256, 0, stream>>>(x, pooled);
    xmean_kernel<<<dim3(4 * 65536 / 256), 256, 0, stream>>>(x, xmean);
    mlp_kernel<<<1, 64, 0, stream>>>(pooled, dsa_w1, dsa_b1, dsa_w2, dsa_b2, kern);
    attn_projg_kernel<<<dim3(65536 / 32, 4), 256, 0, stream>>>(
        x, xmean, kern, gwF, gff_gb, projg);
    fuse_conv_kernel<<<dim3(WW / 8, HH / 4, 4), 256, 0, stream>>>(
        x, projg, lwF, gff_lb, fweight, cwF,
        bn_g, bn_b, bn_m, bn_v, (float*)d_out);
}
